// CrossEncoderLayer_77421080478083
// MI455X (gfx1250) — matrix operand order repaired and checked by reading, not compiled, not run on hardware
//
#include <hip/hip_runtime.h>
#include <hip/hip_bf16.h>

typedef __bf16 bf16_t;
typedef __attribute__((ext_vector_type(8)))  __bf16 v8bf;
typedef __attribute__((ext_vector_type(16))) __bf16 v16bf;
typedef __attribute__((ext_vector_type(8)))  float  v8f;
typedef __attribute__((ext_vector_type(4)))  unsigned int u32x4;
typedef __attribute__((ext_vector_type(8)))  int i32x8;
typedef __attribute__((ext_vector_type(4)))  int i32x4;

#define B_   8
#define NQ_  1024
#define MK_  2048
#define EMB_ 512
#define HID_ 2048
#define NH_  8
#define HD_  64

// ---------- helpers ----------

__device__ __forceinline__ bf16_t f2bf(float f) {
  unsigned u = __builtin_bit_cast(unsigned, f);
  u += 0x7fffu + ((u >> 16) & 1u);           // round-to-nearest-even
  unsigned short s = (unsigned short)(u >> 16);
  return __builtin_bit_cast(bf16_t, s);
}

// build a 16-elem bf16 fragment from two aligned 8-elem LDS chunks
__device__ __forceinline__ v16bf frag_ld(const bf16_t* lo, const bf16_t* hi) {
  v8bf a = *(const v8bf*)lo;
  v8bf b = *(const v8bf*)hi;
  v16bf r;
#pragma unroll
  for (int i = 0; i < 8; ++i) { r[i] = a[i]; r[i + 8] = b[i]; }
  return r;
}

__device__ __forceinline__ v8f wmma_bf16(v16bf a, v16bf b, v8f c) {
  return __builtin_amdgcn_wmma_f32_16x16x32_bf16(false, a, false, b, (short)0, c,
                                                 false, false);
}

// TDM: DMA a 2D bf16 tile (tile1 rows x tile0 elems) from global into LDS,
// inserting pad_amt(+1) DWORDs of padding after each 2^(pad_int+1) DWORDs so
// LDS rows land on our padded tile stride.  D# layout per CDNA5 ISA ch.8.
// (6-arg builtin form: amdgpu-toolchain / clang-23)
__device__ __forceinline__ void tdm_load_2d(unsigned lds_addr, const void* g,
                                            unsigned dim0, unsigned dim1,
                                            unsigned tile0, unsigned tile1,
                                            unsigned long long stride0,
                                            unsigned pad_int, unsigned pad_amt) {
  unsigned long long ga = (unsigned long long)(size_t)g;
  u32x4 g0;
  g0[0] = 1u;                                    // count=1, user descriptor
  g0[1] = lds_addr;                              // LDS byte address
  g0[2] = (unsigned)ga;                          // global_addr[31:0]
  g0[3] = (unsigned)(ga >> 32) | (2u << 30);     // global_addr[56:32] | type=2
  i32x8 g1;
  g1[0] = (int)((1u << 16) |                     // data_size = 2 bytes
                (1u << 20) |                     // pad_enable
                (pad_int << 22) | (pad_amt << 25));
  g1[1] = (int)(dim0 << 16);                     // tensor_dim0[15:0]
  g1[2] = (int)((dim0 >> 16) | (dim1 << 16));    // dim0[31:16] | dim1[15:0]
  g1[3] = (int)((dim1 >> 16) | (tile0 << 16));   // dim1[31:16] | tile_dim0
  g1[4] = (int)(tile1 & 0xffffu);                // tile_dim1 (tile_dim2 = 0)
  g1[5] = (int)(unsigned)stride0;                // tensor_dim0_stride[31:0]
  g1[6] = (int)(unsigned)(stride0 >> 32);        // stride[47:32]
  g1[7] = 0;
  i32x4 z4 = {0, 0, 0, 0};                       // 2D: groups 2/3 unused
  i32x8 z8 = {0, 0, 0, 0, 0, 0, 0, 0};
  __builtin_amdgcn_tensor_load_to_lds(g0, g1, z4, z4, z8, 0);
}

// ---------- weight convert + transpose: W[K][N] f32 -> Wt[N][K] bf16 ----------

__global__ __launch_bounds__(256)
void k_wt(const float* __restrict__ W, bf16_t* __restrict__ Wt, int K, int N) {
  size_t idx = (size_t)blockIdx.x * 256 + threadIdx.x;
  int k = (int)(idx / N);
  int n = (int)(idx % N);
  Wt[(size_t)n * K + k] = f2bf(W[idx]);
}

// ---------- layernorm over 512 cols: f32 in -> bf16 out ----------

__global__ __launch_bounds__(128)
void k_ln(const float* __restrict__ x, const float* __restrict__ g,
          const float* __restrict__ bta, bf16_t* __restrict__ out) {
  int row = blockIdx.x;
  int t = threadIdx.x;
  const float* xr = x + (size_t)row * EMB_;
  float v[4], s = 0.f, s2 = 0.f;
#pragma unroll
  for (int j = 0; j < 4; ++j) {
    v[j] = xr[t + 128 * j];
    s += v[j];
    s2 += v[j] * v[j];
  }
#pragma unroll
  for (int off = 16; off > 0; off >>= 1) {
    s += __shfl_xor(s, off);
    s2 += __shfl_xor(s2, off);
  }
  __shared__ float rs[4], rs2[4];
  if ((t & 31) == 0) { rs[t >> 5] = s; rs2[t >> 5] = s2; }
  __syncthreads();
  s = rs[0] + rs[1] + rs[2] + rs[3];
  s2 = rs2[0] + rs2[1] + rs2[2] + rs2[3];
  float mu = s * (1.f / EMB_);
  float var = s2 * (1.f / EMB_) - mu * mu;
  float rinv = rsqrtf(var + 1e-5f);
  bf16_t* orow = out + (size_t)row * EMB_;
#pragma unroll
  for (int j = 0; j < 4; ++j) {
    int c = t + 128 * j;
    orow[c] = f2bf((v[j] - mu) * rinv * g[c] + bta[c]);
  }
}

// ---------- generic bf16 WMMA GEMM: out = epi(A[M][K] @ Wt[N][K]^T) ----------
// EPI: 0 = +bias -> bf16 ; 1 = +res(f32) -> f32 ; 2 = +bias,GELU -> bf16 ;
//      3 = +bias,+res -> f32

template <int EPI>
__global__ __launch_bounds__(128)
void k_gemm(const bf16_t* __restrict__ A, const bf16_t* __restrict__ Wt,
            const float* __restrict__ bias, const float* __restrict__ res,
            void* __restrict__ outp, int Mr, int Nc, int Kc) {
  __shared__ bf16_t As[64 * 40];   // row stride 40 elems (80B, 16B-aligned)
  __shared__ bf16_t Bs[64 * 40];
  const int t = threadIdx.x;
  const int wv = t >> 5, lane = t & 31, l15 = lane & 15, hh = lane >> 4;
  const int wr = wv >> 1, wc = wv & 1;
  const int m0 = blockIdx.y * 64, n0 = blockIdx.x * 64;

  v8f acc[2][2];
#pragma unroll
  for (int mi = 0; mi < 2; ++mi)
#pragma unroll
    for (int ni = 0; ni < 2; ++ni)
#pragma unroll
      for (int i = 0; i < 8; ++i) acc[mi][ni][i] = 0.f;

  for (int k0 = 0; k0 < Kc; k0 += 32) {
    __syncthreads();                     // prev-iter LDS consumers done
    if (wv == 0) {
      // 64x32 tiles; LDS rows: 16 DW data + 4 DW pad -> encode (3,3)
      tdm_load_2d((unsigned)(size_t)&As[0], &A[(size_t)m0 * Kc + k0],
                  (unsigned)Kc, (unsigned)Mr, 32u, 64u,
                  (unsigned long long)Kc, 3u, 3u);
      tdm_load_2d((unsigned)(size_t)&Bs[0], &Wt[(size_t)n0 * Kc + k0],
                  (unsigned)Kc, (unsigned)Nc, 32u, 64u,
                  (unsigned long long)Kc, 3u, 3u);
      __builtin_amdgcn_s_wait_tensorcnt(0);
    }
    __syncthreads();                     // tiles visible to all waves

    v16bf af[2], bfr[2];
#pragma unroll
    for (int mi = 0; mi < 2; ++mi) {
      const bf16_t* pa = &As[(wr * 32 + mi * 16 + l15) * 40 + hh * 16];
      af[mi] = frag_ld(pa, pa + 8);               // A: K half*16 + 0..15
    }
#pragma unroll
    for (int ni = 0; ni < 2; ++ni) {
      const bf16_t* pb = &Bs[(wc * 32 + ni * 16 + l15) * 40 + hh * 16];
      bfr[ni] = frag_ld(pb, pb + 8);              // B: K half*16 + 0..15
    }
#pragma unroll
    for (int mi = 0; mi < 2; ++mi)
#pragma unroll
      for (int ni = 0; ni < 2; ++ni)
        acc[mi][ni] = wmma_bf16(af[mi], bfr[ni], acc[mi][ni]);
  }

#pragma unroll
  for (int mi = 0; mi < 2; ++mi)
#pragma unroll
    for (int ni = 0; ni < 2; ++ni)
#pragma unroll
      for (int i = 0; i < 8; ++i) {
        int row = m0 + wr * 32 + mi * 16 + 8 * hh + i;
        int col = n0 + wc * 32 + ni * 16 + l15;
        float v = acc[mi][ni][i];
        size_t oidx = (size_t)row * Nc + col;
        if (EPI == 0) {
          v += bias[col];
          ((bf16_t*)outp)[oidx] = f2bf(v);
        } else if (EPI == 1) {
          v += res[oidx];
          ((float*)outp)[oidx] = v;
        } else if (EPI == 2) {
          v += bias[col];
          v = 0.5f * v * (1.f + erff(v * 0.70710678118f));  // exact GELU
          ((bf16_t*)outp)[oidx] = f2bf(v);
        } else {
          v += bias[col] + res[oidx];
          ((float*)outp)[oidx] = v;
        }
      }
}

// ---------- flash attention: per (batch, head, 64-query block) ----------

__global__ __launch_bounds__(128)
void k_attn(const bf16_t* __restrict__ Q, const bf16_t* __restrict__ K,
            const bf16_t* __restrict__ V, bf16_t* __restrict__ O) {
  __shared__ bf16_t Qs[64 * 72];     // [qrow][d]   stride 72 (144B, 16B-aligned)
  __shared__ bf16_t Ks[64 * 72];     // [kvrow][d]  == B^T for S = Q K^T
  __shared__ bf16_t Vs[64 * 72];     // [d][kvrow]  == B^T for O = P V
  __shared__ bf16_t Ps[4 * 16 * 72]; // per-wave 16x64 P tile

  const int t = threadIdx.x;
  const int wv = t >> 5, lane = t & 31, l15 = lane & 15, hh = lane >> 4;
  const int qb0 = blockIdx.x * 64;
  const int h_idx = blockIdx.y;
  const int b = blockIdx.z;
  const int hcol = h_idx * HD_;
  bf16_t* Pw = &Ps[wv * 16 * 72];

  // stage Q block (64 x 64) via TDM; completion covered by the first
  // s_wait_tensorcnt inside the loop (same wave, in-order TENSORcnt)
  if (wv == 0) {
    tdm_load_2d((unsigned)(size_t)&Qs[0],
                &Q[(size_t)(b * NQ_ + qb0) * EMB_ + hcol],
                EMB_, (unsigned)(B_ * NQ_), 64u, 64u,
                (unsigned long long)EMB_, 4u, 3u);  // 32 DW data + 4 DW pad
  }

  float mrow[8], lrow[8];
  v8f oacc[4];
#pragma unroll
  for (int i = 0; i < 8; ++i) { mrow[i] = -3.0e38f; lrow[i] = 0.f; }
#pragma unroll
  for (int jd = 0; jd < 4; ++jd)
#pragma unroll
    for (int i = 0; i < 8; ++i) oacc[jd][i] = 0.f;

  for (int kb = 0; kb < MK_; kb += 64) {
    // stage K chunk via TDM
    if (wv == 0) {
      tdm_load_2d((unsigned)(size_t)&Ks[0],
                  &K[(size_t)(b * MK_ + kb) * EMB_ + hcol],
                  EMB_, (unsigned)(B_ * MK_), 64u, 64u,
                  (unsigned long long)EMB_, 4u, 3u);
    }
    // stage V chunk transposed (TDM cannot transpose): Vs[d][kv]
    {
      int r = t >> 1;
      int c0 = (t & 1) * 32;
      const bf16_t* vrow = &V[(size_t)(b * MK_ + kb + r) * EMB_ + hcol + c0];
#pragma unroll
      for (int jv = 0; jv < 4; ++jv) {
        v8bf vv = *(const v8bf*)&vrow[jv * 8];
#pragma unroll
        for (int e = 0; e < 8; ++e) Vs[(c0 + jv * 8 + e) * 72 + r] = vv[e];
      }
    }
    if (wv == 0) __builtin_amdgcn_s_wait_tensorcnt(0);
    __syncthreads();

    // S = Q K^T for this wave's 16 query rows (all 64 kv cols)
    v8f s[4];
#pragma unroll
    for (int jn = 0; jn < 4; ++jn)
#pragma unroll
      for (int i = 0; i < 8; ++i) s[jn][i] = 0.f;
#pragma unroll
    for (int ks = 0; ks < 2; ++ks) {
      const bf16_t* pa = &Qs[(wv * 16 + l15) * 72 + ks * 32 + hh * 16];
      v16bf aq = frag_ld(pa, pa + 8);
#pragma unroll
      for (int jn = 0; jn < 4; ++jn) {
        const bf16_t* pb = &Ks[(jn * 16 + l15) * 72 + ks * 32 + hh * 16];
        s[jn] = wmma_bf16(aq, frag_ld(pb, pb + 8), s[jn]);
      }
    }

    // online softmax (row = wv*16 + 8*hh + i, shared by 16-lane half-group)
#pragma unroll
    for (int i = 0; i < 8; ++i) {
      float mx = -3.0e38f;
#pragma unroll
      for (int jn = 0; jn < 4; ++jn) {
        float v = s[jn][i] * 0.125f;   // 1/sqrt(64)
        s[jn][i] = v;
        mx = fmaxf(mx, v);
      }
#pragma unroll
      for (int msk = 1; msk < 16; msk <<= 1) mx = fmaxf(mx, __shfl_xor(mx, msk));
      float mnew = fmaxf(mrow[i], mx);
      float corr = __expf(mrow[i] - mnew);
      float psum = 0.f;
#pragma unroll
      for (int jn = 0; jn < 4; ++jn) {
        float pe = __expf(s[jn][i] - mnew);
        s[jn][i] = pe;
        psum += pe;
      }
#pragma unroll
      for (int msk = 1; msk < 16; msk <<= 1) psum += __shfl_xor(psum, msk);
      mrow[i] = mnew;
      lrow[i] = lrow[i] * corr + psum;
#pragma unroll
      for (int jd = 0; jd < 4; ++jd) oacc[jd][i] *= corr;
      // spill P (C-layout) into per-wave LDS in A-tile layout order
#pragma unroll
      for (int jn = 0; jn < 4; ++jn)
        Pw[(i + 8 * hh) * 72 + jn * 16 + l15] = f2bf(s[jn][i]);
    }

    // O += P @ V  (same-wave LDS RAW is in-order)
#pragma unroll
    for (int ks = 0; ks < 2; ++ks) {
      const bf16_t* pa = &Pw[l15 * 72 + ks * 32 + hh * 16];
      v16bf ap = frag_ld(pa, pa + 8);
#pragma unroll
      for (int jd = 0; jd < 4; ++jd) {
        const bf16_t* pb = &Vs[(jd * 16 + l15) * 72 + ks * 32 + hh * 16];
        oacc[jd] = wmma_bf16(ap, frag_ld(pb, pb + 8), oacc[jd]);
      }
    }
    __syncthreads();
  }

  // normalize and write out
#pragma unroll
  for (int i = 0; i < 8; ++i) {
    float inv = 1.f / lrow[i];
    int grow = b * NQ_ + qb0 + wv * 16 + 8 * hh + i;
#pragma unroll
    for (int jd = 0; jd < 4; ++jd) {
      int col = hcol + jd * 16 + l15;
      O[(size_t)grow * EMB_ + col] = f2bf(oacc[jd][i] * inv);
    }
  }
}

// ---------- driver ----------

extern "C" void kernel_launch(void* const* d_in, const int* in_sizes, int n_in,
                              void* d_out, int out_size, void* d_ws,
                              size_t ws_size, hipStream_t stream) {
  const float* xq   = (const float*)d_in[0];
  const float* xk   = (const float*)d_in[1];
  const float* xv   = (const float*)d_in[2];
  const float* Wq   = (const float*)d_in[3];
  const float* bq   = (const float*)d_in[4];
  const float* Wk   = (const float*)d_in[5];
  const float* bk   = (const float*)d_in[6];
  const float* Wv   = (const float*)d_in[7];
  const float* bv   = (const float*)d_in[8];
  const float* Wo   = (const float*)d_in[9];
  const float* g1   = (const float*)d_in[10];
  const float* b1   = (const float*)d_in[11];
  const float* g2   = (const float*)d_in[12];
  const float* b2   = (const float*)d_in[13];
  const float* Wff1 = (const float*)d_in[14];
  const float* bff1 = (const float*)d_in[15];
  const float* Wff2 = (const float*)d_in[16];
  const float* bff2 = (const float*)d_in[17];

  const int RQ = B_ * NQ_;   // 8192
  const int RK = B_ * MK_;   // 16384

  char* wsb = (char*)d_ws;
  size_t off = 0;
  auto carve = [&](size_t bytes) -> void* {
    void* p = wsb + off;
    off = (off + bytes + 255) & ~(size_t)255;
    return p;
  };
  bf16_t* Wtq  = (bf16_t*)carve((size_t)EMB_ * EMB_ * 2);
  bf16_t* Wtk  = (bf16_t*)carve((size_t)EMB_ * EMB_ * 2);
  bf16_t* Wtv  = (bf16_t*)carve((size_t)EMB_ * EMB_ * 2);
  bf16_t* Wto  = (bf16_t*)carve((size_t)EMB_ * EMB_ * 2);
  bf16_t* Wtf1 = (bf16_t*)carve((size_t)EMB_ * HID_ * 2);
  bf16_t* Wtf2 = (bf16_t*)carve((size_t)HID_ * EMB_ * 2);
  bf16_t* nq   = (bf16_t*)carve((size_t)RQ * EMB_ * 2);
  bf16_t* nk   = (bf16_t*)carve((size_t)RK * EMB_ * 2);
  bf16_t* nv   = (bf16_t*)carve((size_t)RK * EMB_ * 2);
  bf16_t* qb   = (bf16_t*)carve((size_t)RQ * EMB_ * 2);
  bf16_t* kb   = (bf16_t*)carve((size_t)RK * EMB_ * 2);
  bf16_t* vb   = (bf16_t*)carve((size_t)RK * EMB_ * 2);
  bf16_t* attn = (bf16_t*)carve((size_t)RQ * EMB_ * 2);
  float*  avec = (float*) carve((size_t)RQ * EMB_ * 4);
  bf16_t* na   = (bf16_t*)carve((size_t)RQ * EMB_ * 2);
  bf16_t* hbuf = (bf16_t*)carve((size_t)RQ * HID_ * 2);

  // weight prep (bf16, transposed)
  k_wt<<<1024, 256, 0, stream>>>(Wq, Wtq, EMB_, EMB_);
  k_wt<<<1024, 256, 0, stream>>>(Wk, Wtk, EMB_, EMB_);
  k_wt<<<1024, 256, 0, stream>>>(Wv, Wtv, EMB_, EMB_);
  k_wt<<<1024, 256, 0, stream>>>(Wo, Wto, EMB_, EMB_);
  k_wt<<<4096, 256, 0, stream>>>(Wff1, Wtf1, EMB_, HID_);
  k_wt<<<4096, 256, 0, stream>>>(Wff2, Wtf2, HID_, EMB_);

  // pre-norm
  k_ln<<<RQ, 128, 0, stream>>>(xq, g1, b1, nq);
  k_ln<<<RK, 128, 0, stream>>>(xk, g1, b1, nk);
  k_ln<<<RK, 128, 0, stream>>>(xv, g1, b1, nv);

  // Q/K/V projections (bias -> bf16)
  k_gemm<0><<<dim3(EMB_ / 64, RQ / 64), 128, 0, stream>>>(nq, Wtq, bq, nullptr,
                                                          qb, RQ, EMB_, EMB_);
  k_gemm<0><<<dim3(EMB_ / 64, RK / 64), 128, 0, stream>>>(nk, Wtk, bk, nullptr,
                                                          kb, RK, EMB_, EMB_);
  k_gemm<0><<<dim3(EMB_ / 64, RK / 64), 128, 0, stream>>>(nv, Wtv, bv, nullptr,
                                                          vb, RK, EMB_, EMB_);

  // attention
  k_attn<<<dim3(NQ_ / 64, NH_, B_), 128, 0, stream>>>(qb, kb, vb, attn);

  // output projection + residual (f32)
  k_gemm<1><<<dim3(EMB_ / 64, RQ / 64), 128, 0, stream>>>(
      attn, Wto, nullptr, xq, avec, RQ, EMB_, EMB_);

  // FFN
  k_ln<<<RQ, 128, 0, stream>>>(avec, g2, b2, na);
  k_gemm<2><<<dim3(HID_ / 64, RQ / 64), 128, 0, stream>>>(
      na, Wtf1, bff1, nullptr, hbuf, RQ, HID_, EMB_);
  k_gemm<3><<<dim3(EMB_ / 64, RQ / 64), 128, 0, stream>>>(
      hbuf, Wtf2, bff2, avec, (float*)d_out, RQ, EMB_, HID_);
}
